// DozerAttentionLayer_19653770346744
// MI455X (gfx1250) — matrix operand order repaired and checked by reading, not compiled, not run on hardware
//
#include <hip/hip_runtime.h>

// ---------------------------------------------------------------------------
// Problem constants (fixed by the reference):
//   B=2, L=S=2048, D=1024, H=16, DK=64
// ---------------------------------------------------------------------------
#define CB  2
#define CL  2048
#define CS  2048
#define CD  1024
#define CH  16
#define CDK 64

typedef __bf16       bf16x16 __attribute__((ext_vector_type(16)));
typedef float        f32x8   __attribute__((ext_vector_type(8)));
typedef float        f32x4   __attribute__((ext_vector_type(4)));
typedef unsigned int u32x4   __attribute__((ext_vector_type(4)));
typedef unsigned int uint32x4 __attribute__((ext_vector_type(4)));
typedef int          int32x4 __attribute__((ext_vector_type(4)));
typedef int          int32x8 __attribute__((ext_vector_type(8)));

union Frag16 {            // 16 bf16 values = 8 VGPRs
    bf16x16 v;
    u32x4   q[2];
};

__device__ __forceinline__ unsigned short f2bf(float f) {
    unsigned int u = __float_as_uint(f);
    unsigned int r = u + 0x7FFFu + ((u >> 16) & 1u);   // round-to-nearest-even
    return (unsigned short)(r >> 16);
}

// ---------------------------------------------------------------------------
// Tensor Data Mover: load a 2D tile of 2-byte elements into LDS.
//   D# group0: count=1 | lds_addr | global_addr(57b) | type=2 ("image")
//   D# group1: data_size=1 (2B), tile_dim0=row_elems, tile_dim1=rows,
//              tensor_dim0_stride=stride_elems (row-to-row, elements)
// Rows land back-to-back in LDS: LDS[row*row_elems + x], i.e. [rows][row_elems].
// Issued once per wave (EXEC ignored); completion via TENSORcnt.
// NOTE: this toolchain's builtin takes 6 args (g0, g1, g2, g3, g4, cpol).
// ---------------------------------------------------------------------------
__device__ __forceinline__ void tdm_load_tile_b16(
    unsigned lds_byte_off, const void* gptr,
    unsigned row_elems, unsigned rows, unsigned stride_elems)
{
    unsigned long long ga = (unsigned long long)(uintptr_t)gptr;
    uint32x4 g0;
    g0[0] = 1u;                                        // count=1, user D#
    g0[1] = lds_byte_off;                              // lds_addr (bytes)
    g0[2] = (unsigned)ga;                              // global_addr[31:0]
    g0[3] = (unsigned)(ga >> 32) | 0x80000000u;        // addr[56:32] | type=2

    const unsigned td0 = stride_elems;                 // tensor_dim0 >= tile row
    const unsigned td1 = 0x7FFFFFFFu;                  // tensor_dim1: no clip
    int32x8 g1;
    g1[0] = 0x00010000;                                // data_size=1 (2 bytes)
    g1[1] = (int)((td0 & 0xFFFFu) << 16);              // [atomic_addr=0 | td0.lo]
    g1[2] = (int)((td0 >> 16) | ((td1 & 0xFFFFu) << 16));
    g1[3] = (int)((td1 >> 16) | (row_elems << 16));    // td1.hi | tile_dim0
    g1[4] = (int)rows;                                 // tile_dim1 (tile_dim2=0)
    g1[5] = (int)stride_elems;                         // tensor_dim0_stride lo
    g1[6] = 0;                                         // stride hi | dim1_stride
    g1[7] = 0;

    int32x4 z4 = {};
    int32x8 z8 = {};
    __builtin_amdgcn_tensor_load_to_lds(g0, g1, z4, z4, z8, 0);
}

// ---------------------------------------------------------------------------
// Wave-level MMA on staged LDS tiles (one WMMA K-step of 32).
// A-fragment (16-bit A 16x32, ISA 7.12.2): row M = lane&15, K-half by lane>=16,
// two ds_load_b128. B-fragment: column N = lane&15, same K order as A (gathered).
// LDA/LDB are the LDS row strides (in halves) of the A / B tiles.
// ---------------------------------------------------------------------------
template<int LDA, int LDB, int MT, int NT>
__device__ __forceinline__ void tile_mma(const unsigned short* lA,
                                         const unsigned short* lB,
                                         int wm0, int wn0,
                                         f32x8 (&acc)[MT][NT])
{
    const int lane = threadIdx.x & 31;
    const int am   = lane & 15;
    const int kh   = (lane >> 4) * 8;

    Frag16 af[MT], bfr[NT];
#pragma unroll
    for (int i = 0; i < MT; ++i) {
        const unsigned short* p = lA + (wm0 + i * 16 + am) * LDA + kh;
        af[i].q[0] = *(const u32x4*)(p);
        af[i].q[1] = *(const u32x4*)(p + 16);
    }
#pragma unroll
    for (int j = 0; j < NT; ++j) {
        const unsigned short* p = lB + wn0 + j * 16 + am;
#pragma unroll
        for (int s = 0; s < 16; ++s) {
            const int k = kh + (s & 7) + ((s >> 3) << 4);
            bfr[j].v[s] = __builtin_bit_cast(__bf16, p[k * LDB]);
        }
    }
#pragma unroll
    for (int i = 0; i < MT; ++i)
#pragma unroll
        for (int j = 0; j < NT; ++j)
            acc[i][j] = __builtin_amdgcn_wmma_f32_16x16x32_bf16(
                false, af[i].v, false, bfr[j].v, (short)0, acc[i][j],
                false, false);
}

// ---------------------------------------------------------------------------
// Generic dense GEMM:  C[M,N] = A[M,K] * B[K,N] + bias
//   AT = float (convert to bf16 while staging) or unsigned short (raw bf16)
//   OMODE: 0 = fp32 out, 1 = bf16 out, 2 = bf16 out transposed per batch
//          (K-projection: row m = b*S+s, store Kt[(b*CD + col)*CS + s])
// Block tile 128x128x32, 8 waves as 2x4 -> 64x32 per wave (4x2 WMMA accums).
// ---------------------------------------------------------------------------
enum { OUT_F32 = 0, OUT_BF16 = 1, OUT_BF16_KT = 2 };

template<typename AT, int OMODE>
__global__ __launch_bounds__(256) void gemm_wmma(
    const AT* __restrict__ A, int lda,
    const float* __restrict__ Bm, int ldb,
    const float* __restrict__ bias,
    void* __restrict__ Cptr, int ldc,
    int M, int N, int K)
{
    constexpr int BM = 128, BN = 128, BK = 32;
    __shared__ alignas(16) unsigned short lA[BM * BK];
    __shared__ alignas(16) unsigned short lB[BK * BN];

    const int tid  = threadIdx.x;
    const int wave = tid >> 5;
    const int wm0  = (wave >> 2) * 64;
    const int wn0  = (wave & 3) * 32;
    const int bm0  = blockIdx.y * BM;
    const int bn0  = blockIdx.x * BN;

    f32x8 zero = {};
    f32x8 acc[4][2];
#pragma unroll
    for (int i = 0; i < 4; ++i)
#pragma unroll
        for (int j = 0; j < 2; ++j) acc[i][j] = zero;

    for (int k0 = 0; k0 < K; k0 += BK) {
        if (k0 + BK < K) {   // hint next tiles toward cache
            __builtin_prefetch(A  + (size_t)(bm0 + (tid >> 1)) * lda + k0 + BK, 0, 1);
            __builtin_prefetch(Bm + (size_t)(k0 + BK + (tid >> 3)) * ldb + bn0, 0, 1);
        }
        // ---- stage A (BM x BK) ----
        if constexpr (sizeof(AT) == 4) {           // fp32 -> bf16
#pragma unroll
            for (int i = 0; i < 4; ++i) {
                int slot = tid + i * 256;          // 1024 float4 slots
                int r = slot >> 3;
                int c = (slot & 7) * 4;
                f32x4 v = *(const f32x4*)((const float*)A + (size_t)(bm0 + r) * lda + k0 + c);
                unsigned short* d = lA + r * BK + c;
                d[0] = f2bf(v[0]); d[1] = f2bf(v[1]);
                d[2] = f2bf(v[2]); d[3] = f2bf(v[3]);
            }
        } else {                                   // already bf16
#pragma unroll
            for (int i = 0; i < 2; ++i) {
                int slot = tid + i * 256;          // 512 x 8-half slots
                int r = slot >> 2;
                int c = (slot & 3) * 8;
                *(u32x4*)(lA + r * BK + c) =
                    *(const u32x4*)((const unsigned short*)A + (size_t)(bm0 + r) * lda + k0 + c);
            }
        }
        // ---- stage B (BK x BN), fp32 -> bf16 ----
#pragma unroll
        for (int i = 0; i < 4; ++i) {
            int slot = tid + i * 256;              // 1024 float4 slots
            int r = slot >> 5;
            int c = (slot & 31) * 4;
            f32x4 v = *(const f32x4*)(Bm + (size_t)(k0 + r) * ldb + bn0 + c);
            unsigned short* d = lB + r * BN + c;
            d[0] = f2bf(v[0]); d[1] = f2bf(v[1]);
            d[2] = f2bf(v[2]); d[3] = f2bf(v[3]);
        }
        __syncthreads();
        tile_mma<BK, BN, 4, 2>(lA, lB, wm0, wn0, acc);
        __syncthreads();
    }

    // ---- epilogue: bias + store ----
    const int lane  = tid & 31;
    const int rbase = (lane >> 4) * 8;
    const int cc    = lane & 15;
#pragma unroll
    for (int i = 0; i < 4; ++i)
#pragma unroll
        for (int j = 0; j < 2; ++j) {
            int col = bn0 + wn0 + j * 16 + cc;
            float bv = bias[col];
#pragma unroll
            for (int r = 0; r < 8; ++r) {
                int row = bm0 + wm0 + i * 16 + rbase + r;
                float val = acc[i][j][r] + bv;
                if constexpr (OMODE == OUT_F32) {
                    ((float*)Cptr)[(size_t)row * ldc + col] = val;
                } else if constexpr (OMODE == OUT_BF16) {
                    ((unsigned short*)Cptr)[(size_t)row * ldc + col] = f2bf(val);
                } else {                           // K transposed per batch
                    int bb = row >> 11;            // row = b*CS + s
                    int s  = row & (CS - 1);
                    ((unsigned short*)Cptr)[((size_t)bb * CD + col) * CS + s] = f2bf(val);
                }
            }
        }
}

// ---------------------------------------------------------------------------
// scores[b,h,l,s] = scale * sum_d q[b,l,h,d] * kt[b, h*DK+d, s]  (pre-softmax)
// Single staging of the whole DK=64 via two TDM tile loads (pure bf16 copies):
//   lA[128][64] <- Q head rows (row stride D), lB[64][128] <- Kt rows (stride S)
// then two WMMA K-steps. LDS use: 32 KB of 320 KB.
// ---------------------------------------------------------------------------
__global__ __launch_bounds__(256) void attn_scores(
    const unsigned short* __restrict__ Qb,
    const unsigned short* __restrict__ Kt,
    float* __restrict__ attn, float scale)
{
    constexpr int BM = 128, BN = 128;
    __shared__ alignas(16) unsigned short lA[BM * CDK];   // 128 x 64
    __shared__ alignas(16) unsigned short lB[CDK * BN];   // 64 x 128

    const int tid  = threadIdx.x;
    const int wave = tid >> 5;
    const int wm0  = (wave >> 2) * 64;
    const int wn0  = (wave & 3) * 32;
    const int bh   = blockIdx.z;
    const int b    = bh >> 4;
    const int h    = bh & 15;
    const int l0   = blockIdx.y * BM;
    const int s0   = blockIdx.x * BN;

    const unsigned short* qtile = Qb + (size_t)(b * CL + l0) * CD + h * CDK;
    const unsigned short* ktile = Kt + ((size_t)b * CD + h * CDK) * CS + s0;

    if (wave == 0) {
        tdm_load_tile_b16((unsigned)(uintptr_t)lA, qtile, CDK, BM, CD);
        tdm_load_tile_b16((unsigned)(uintptr_t)lB, ktile, BN, CDK, CS);
        __builtin_amdgcn_s_wait_tensorcnt(0);
    }
    __syncthreads();

    f32x8 zero = {};
    f32x8 acc[4][2];
#pragma unroll
    for (int i = 0; i < 4; ++i)
#pragma unroll
        for (int j = 0; j < 2; ++j) acc[i][j] = zero;

#pragma unroll
    for (int kk = 0; kk < CDK; kk += 32)
        tile_mma<CDK, BN, 4, 2>(lA + kk, lB + kk * BN, wm0, wn0, acc);

    const int lane  = tid & 31;
    const int rbase = (lane >> 4) * 8;
    const int cc    = lane & 15;
    float* arow = attn + (size_t)bh * CL * CS;
#pragma unroll
    for (int i = 0; i < 4; ++i)
#pragma unroll
        for (int j = 0; j < 2; ++j)
#pragma unroll
            for (int r = 0; r < 8; ++r) {
                int row = l0 + wm0 + i * 16 + rbase + r;
                int col = s0 + wn0 + j * 16 + cc;
                arow[(size_t)row * CS + col] = acc[i][j][r] * scale;
            }
}

// ---------------------------------------------------------------------------
// In-place masked row softmax over attn[b,h,l,:]  (row length S=2048).
// One 256-thread block (8 wave32) per row; 8 elements/thread.
// ---------------------------------------------------------------------------
__global__ __launch_bounds__(256) void softmax_rows(
    float* __restrict__ attn, const unsigned char* __restrict__ mask)
{
    const int row = blockIdx.x;           // (b*H + h)*L + l
    const int l   = row & (CL - 1);
    float* p = attn + (size_t)row * CS;
    const unsigned char* mrow = mask + (size_t)l * CS;
    const int tid = threadIdx.x;

    const float NEG = -3.402823466e38f;   // finfo(float32).min
    float v[8];
    float mx = NEG;
#pragma unroll
    for (int i = 0; i < 8; ++i) {
        int idx = tid + i * 256;
        float x = p[idx];
        v[i] = mrow[idx] ? x : NEG;
        mx = fmaxf(mx, v[i]);
    }
#pragma unroll
    for (int off = 16; off > 0; off >>= 1)
        mx = fmaxf(mx, __shfl_xor(mx, off, 32));

    __shared__ float red[8];
    const int wave = tid >> 5, lane = tid & 31;
    if (lane == 0) red[wave] = mx;
    __syncthreads();
    mx = red[0];
#pragma unroll
    for (int w = 1; w < 8; ++w) mx = fmaxf(mx, red[w]);

    float sum = 0.f;
#pragma unroll
    for (int i = 0; i < 8; ++i) {
        float e = __expf(v[i] - mx);
        v[i] = e;
        sum += e;
    }
#pragma unroll
    for (int off = 16; off > 0; off >>= 1)
        sum += __shfl_xor(sum, off, 32);
    __syncthreads();
    if (lane == 0) red[wave] = sum;
    __syncthreads();
    sum = red[0];
#pragma unroll
    for (int w = 1; w < 8; ++w) sum += red[w];

    float inv = 1.0f / sum;
#pragma unroll
    for (int i = 0; i < 8; ++i) p[tid + i * 256] = v[i] * inv;
}

// ---------------------------------------------------------------------------
// ctx[b,l,h,:] = attn[b,h,l,:] @ v[b,:,h,:]
// Block tile 128x64x32, 8 waves as 4x2 -> 32x32 per wave (2x2 accums).
// ---------------------------------------------------------------------------
__global__ __launch_bounds__(256) void attn_v(
    const float* __restrict__ attn,
    const unsigned short* __restrict__ Vb,
    unsigned short* __restrict__ Ctx)
{
    constexpr int BM = 128, BN = 64, BK = 32;
    __shared__ alignas(16) unsigned short lA[BM * BK];
    __shared__ alignas(16) unsigned short lB[BK * BN];

    const int tid  = threadIdx.x;
    const int wave = tid >> 5;
    const int wm0  = (wave >> 1) * 32;
    const int wn0  = (wave & 1) * 32;
    const int bh   = blockIdx.z;
    const int b    = bh >> 4;
    const int h    = bh & 15;
    const int l0   = blockIdx.y * BM;
    const float* Arow = attn + (size_t)bh * CL * CS;
    const size_t vbase = (size_t)b * CS * CD + (size_t)h * CDK;

    f32x8 zero = {};
    f32x8 acc[2][2];
#pragma unroll
    for (int i = 0; i < 2; ++i)
#pragma unroll
        for (int j = 0; j < 2; ++j) acc[i][j] = zero;

    for (int k0 = 0; k0 < CS; k0 += BK) {
        // A tile: 128x32 fp32 attn -> bf16
#pragma unroll
        for (int i = 0; i < 4; ++i) {
            int slot = tid + i * 256;
            int r = slot >> 3;
            int c = (slot & 7) * 4;
            f32x4 x = *(const f32x4*)(Arow + (size_t)(l0 + r) * CS + k0 + c);
            unsigned short* d = lA + r * BK + c;
            d[0] = f2bf(x[0]); d[1] = f2bf(x[1]);
            d[2] = f2bf(x[2]); d[3] = f2bf(x[3]);
        }
        // B tile: 32 rows x 64 contiguous halves of V head
        {
            int r = tid >> 3;
            int c = (tid & 7) * 8;
            *(u32x4*)(lB + r * BN + c) =
                *(const u32x4*)(Vb + vbase + (size_t)(k0 + r) * CD + c);
        }
        __syncthreads();
        tile_mma<BK, BN, 2, 2>(lA, lB, wm0, wn0, acc);
        __syncthreads();
    }

    const int lane  = tid & 31;
    const int rbase = (lane >> 4) * 8;
    const int cc    = lane & 15;
#pragma unroll
    for (int i = 0; i < 2; ++i)
#pragma unroll
        for (int j = 0; j < 2; ++j)
#pragma unroll
            for (int r = 0; r < 8; ++r) {
                int row = l0 + wm0 + i * 16 + rbase + r;
                int col = wn0 + j * 16 + cc;
                Ctx[((size_t)b * CL + row) * CD + h * CDK + col] = f2bf(acc[i][j][r]);
            }
}

// ---------------------------------------------------------------------------
// kernel_launch
// ---------------------------------------------------------------------------
extern "C" void kernel_launch(void* const* d_in, const int* in_sizes, int n_in,
                              void* d_out, int out_size, void* d_ws, size_t ws_size,
                              hipStream_t stream)
{
    const float* queries = (const float*)d_in[0];
    const float* keys    = (const float*)d_in[1];
    const float* values  = (const float*)d_in[2];
    const unsigned char* mask = (const unsigned char*)d_in[3];
    const float* Wq = (const float*)d_in[4];
    const float* bq = (const float*)d_in[5];
    const float* Wk = (const float*)d_in[6];
    const float* bk = (const float*)d_in[7];
    const float* Wv = (const float*)d_in[8];
    const float* bv = (const float*)d_in[9];
    const float* Wo = (const float*)d_in[10];
    const float* bo = (const float*)d_in[11];

    const size_t TOK = (size_t)CB * CL;        // 4096 rows
    const size_t QKV = TOK * CD;               // 4,194,304 elements

    float* out  = (float*)d_out;               // [B, L, D]
    float* attn = out + QKV;                   // [B, H, L, S]

    unsigned short* qb  = (unsigned short*)d_ws;   // bf16 workspace
    unsigned short* ktr = qb + QKV;                // K transposed [B, H*DK, S]
    unsigned short* vb  = ktr + QKV;
    unsigned short* ctx = vb + QKV;

    dim3 blk(256);

    // 1) QKV projections (fp32 in, bf16 out; K written transposed per batch)
    dim3 gproj(CD / 128, (unsigned)(TOK / 128));
    gemm_wmma<float, OUT_BF16><<<gproj, blk, 0, stream>>>(queries, CD, Wq, CD, bq, qb, CD,
                                                          (int)TOK, CD, CD);
    gemm_wmma<float, OUT_BF16_KT><<<gproj, blk, 0, stream>>>(keys, CD, Wk, CD, bk, ktr, CS,
                                                             (int)TOK, CD, CD);
    gemm_wmma<float, OUT_BF16><<<gproj, blk, 0, stream>>>(values,  CD, Wv, CD, bv, vb, CD,
                                                          (int)TOK, CD, CD);

    // 2) scores = scale * Q Kt  -> attn region of d_out (pre-softmax, TDM staged)
    dim3 gsc(CS / 128, CL / 128, CB * CH);
    attn_scores<<<gsc, blk, 0, stream>>>(qb, ktr, attn, 0.125f);

    // 3) masked softmax in place
    softmax_rows<<<dim3(CB * CH * CL), blk, 0, stream>>>(attn, mask);

    // 4) ctx = attn @ V  (bf16 out)
    dim3 gav(1, CL / 128, CB * CH);
    attn_v<<<gav, blk, 0, stream>>>(attn, vb, ctx);

    // 5) out = ctx @ Wo + bo  (fp32 out)
    gemm_wmma<unsigned short, OUT_F32><<<gproj, blk, 0, stream>>>(ctx, CD, Wo, CD, bo, out, CD,
                                                                  (int)TOK, CD, CD);
}
